// SelfAttention_28724741275986
// MI455X (gfx1250) — matrix operand order repaired and checked by reading, not compiled, not run on hardware
//
#include <hip/hip_runtime.h>
#include <cstdint>
#include <cstddef>

// Problem constants (match reference)
#define Bn   2
#define Sn   2048
#define DIMn 1024
#define Hn   16
#define HDn  64
#define Mr   (Bn * Sn)          // 4096 rows for all GEMMs

typedef __bf16 v16bf __attribute__((ext_vector_type(16)));
typedef __bf16 v8bf  __attribute__((ext_vector_type(8)));
typedef __bf16 v4bf  __attribute__((ext_vector_type(4)));
typedef float  v8f   __attribute__((ext_vector_type(8)));
typedef int    v4i   __attribute__((ext_vector_type(4)));

__device__ __forceinline__ v16bf cat8(v8bf lo, v8bf hi) {
  return __builtin_shufflevector(lo, hi, 0,1,2,3,4,5,6,7,8,9,10,11,12,13,14,15);
}
__device__ __forceinline__ v8f wmma_bf16(v16bf a, v16bf b, v8f c) {
  return __builtin_amdgcn_wmma_f32_16x16x32_bf16(false, a, false, b, (short)0, c,
                                                 false, false);
}

// CDNA5 async global->LDS copy (ASYNCcnt path). Builtin exists on this
// toolchain; parameter type is v4i* in global(1)/local(3) address spaces.
#if __has_builtin(__builtin_amdgcn_global_load_async_to_lds_b128)
#define HAVE_ASYNC_LDS 1
#define ASYNC_CP16(gptr, lptr)                                              \
  __builtin_amdgcn_global_load_async_to_lds_b128(                           \
      (__attribute__((address_space(1))) v4i*)(v4i*)(gptr),                 \
      (__attribute__((address_space(3))) v4i*)(v4i*)(lptr), 0, 0)
#else
#define HAVE_ASYNC_LDS 0
#define ASYNC_CP16(gptr, lptr)                                              \
  do { *(v8bf*)(lptr) = *(const v8bf*)(gptr); } while (0)
#endif

// ---------------------------------------------------------------------------
// f32 -> bf16 pack (one-time precision conversion; 4 elems/thread)
// ---------------------------------------------------------------------------
__global__ __launch_bounds__(256)
void pack_bf16_kernel(const float* __restrict__ src, __bf16* __restrict__ dst) {
  const int i = (blockIdx.x * blockDim.x + threadIdx.x) * 4;
  float4 f = *(const float4*)(src + i);
  v4bf o;
  o[0] = (__bf16)f.x; o[1] = (__bf16)f.y; o[2] = (__bf16)f.z; o[3] = (__bf16)f.w;
  *(v4bf*)(dst + i) = o;
}

// ---------------------------------------------------------------------------
// GEMM: C[M,N] = A[M,K] * W[N,K]^T, bf16 in / f32 out, bf16 WMMA math.
// Block = 256 threads = 8 waves; tile 128(M) x 64(N), K-step 32.
// Wave grid 4x2; each wave owns a 32x32 output patch (2x2 WMMA tiles).
// Tiles staged into LDS with GLOBAL_LOAD_ASYNC_TO_LDS_B128 (no VGPR bounce).
// ---------------------------------------------------------------------------
#define BM 128
#define BN 64
#define BK 32

__global__ __launch_bounds__(256)
void gemm_bf16_kernel(const __bf16* __restrict__ A, const __bf16* __restrict__ W,
                      float* __restrict__ C, int M, int N, int K) {
  __shared__ __bf16 As[BM][BK];   // 8 KB
  __shared__ __bf16 Bs[BN][BK];   // 4 KB

  const int nb   = N / BN;
  const int bm   = blockIdx.x / nb;
  const int bn   = blockIdx.x % nb;
  const int tid  = threadIdx.x;
  const int lane = tid & 31;
  const int wave = tid >> 5;
  const int wm   = wave >> 1;       // 0..3 (M direction)
  const int wn   = wave & 1;        // 0..1 (N direction)
  const int g    = lane >> 4;       // half-wave group
  const int ln   = lane & 15;

  // Per-thread staging sources (advance by BK each iteration).
  const __bf16* ga = A + (size_t)(bm * BM + (tid >> 1)) * K + (tid & 1) * 16;
  const __bf16* gb = W + (size_t)(bn * BN + (tid >> 2)) * K + (tid & 3) * 8;
  __bf16* la0 = &As[tid >> 1][(tid & 1) * 16];
  __bf16* lb0 = &Bs[tid >> 2][(tid & 3) * 8];

  v8f acc[2][2] = {};

  for (int k0 = 0; k0 < K; k0 += BK) {
    // --- stage A (128x32 bf16 = 8KB: 2 x b128/thread) and B (4KB: 1 x b128) ---
    ASYNC_CP16(ga + k0,     la0);
    ASYNC_CP16(ga + k0 + 8, la0 + 8);
    ASYNC_CP16(gb + k0,     lb0);
    // prefetch next K-tile (global_prefetch_b8)
    if (k0 + BK < K) {
      __builtin_prefetch(ga + k0 + BK, 0, 0);
      __builtin_prefetch(gb + k0 + BK, 0, 0);
    }
#if HAVE_ASYNC_LDS
    asm volatile("s_wait_asynccnt 0x0" ::: "memory");
#endif
    __syncthreads();

    // --- A fragments: 16-bit 16x32 layout. lane ln = row M; K chunks:
    //     lo half -> K in [8g, 8g+8), hi half -> K in [16+8g, 16+8g+8) ---
    v16bf af[2];
#pragma unroll
    for (int sm = 0; sm < 2; ++sm) {
      const __bf16* rp = &As[wm * 32 + sm * 16 + ln][0];
      af[sm] = cat8(*(const v8bf*)(rp + 8 * g), *(const v8bf*)(rp + 16 + 8 * g));
    }
    // --- B fragments: 32x16 layout. lane ln = column N; K in [8g, 8g+8) and [16+8g, 16+8g+8) ---
    v16bf bfm[2];
#pragma unroll
    for (int sn = 0; sn < 2; ++sn) {
      const __bf16* rp = &Bs[wn * 32 + sn * 16 + ln][0];
      bfm[sn] = cat8(*(const v8bf*)(rp + 8 * g), *(const v8bf*)(rp + 16 + 8 * g));
    }
#pragma unroll
    for (int sm = 0; sm < 2; ++sm)
#pragma unroll
      for (int sn = 0; sn < 2; ++sn)
        acc[sm][sn] = wmma_bf16(af[sm], bfm[sn], acc[sm][sn]);

    __syncthreads();
  }

  // --- write C: reg r -> M = r + 8g, lane ln -> N ---
#pragma unroll
  for (int sm = 0; sm < 2; ++sm)
#pragma unroll
    for (int sn = 0; sn < 2; ++sn) {
      const int gm0 = bm * BM + wm * 32 + sm * 16 + 8 * g;
      const int gn  = bn * BN + wn * 32 + sn * 16 + ln;
#pragma unroll
      for (int r = 0; r < 8; ++r)
        C[(size_t)(gm0 + r) * N + gn] = acc[sm][sn][r];
    }
}

// ---------------------------------------------------------------------------
// RoPE + repack: q,k,v are (B,S,DIM) f32 from the projection GEMMs.
// Emits head-major bf16:
//   qh,kh: (B,H,S,HD)  (q pre-scaled by 1/sqrt(HD), rope applied to q,k)
//   vT:    (B,H,HD,S)  (transposed so PV B-fragments are s-contiguous)
// One thread per (b,h,s,i) rotation pair, i in [0,32).
// ---------------------------------------------------------------------------
__global__ __launch_bounds__(256)
void rope_pack_kernel(const float* __restrict__ q, const float* __restrict__ k,
                      const float* __restrict__ v,
                      const float* __restrict__ fcos, const float* __restrict__ fsin,
                      __bf16* __restrict__ qh, __bf16* __restrict__ kh,
                      __bf16* __restrict__ vT) {
  const int tid = blockIdx.x * blockDim.x + threadIdx.x;   // B*H*S*32 threads
  const int i  = tid & 31;
  const int s  = (tid >> 5) & (Sn - 1);
  const int bh = tid >> 16;              // 5 (i) + 11 (s) bits
  const int h  = bh & (Hn - 1);

  const float c  = fcos[s * 32 + i];
  const float sn = fsin[s * 32 + i];

  const size_t src = ((size_t)(bh >> 4) * Sn + s) * DIMn + h * HDn + 2 * i;
  const size_t dst = ((size_t)bh * Sn + s) * HDn + 2 * i;

  // Q (scaled by 1/sqrt(HD) = 0.125)
  {
    float xr = q[src], xi = q[src + 1];
    qh[dst]     = (__bf16)(0.125f * (xr * c - xi * sn));
    qh[dst + 1] = (__bf16)(0.125f * (xr * sn + xi * c));
  }
  // K
  {
    float xr = k[src], xi = k[src + 1];
    kh[dst]     = (__bf16)(xr * c - xi * sn);
    kh[dst + 1] = (__bf16)(xr * sn + xi * c);
  }
  // V transpose: vT[b][h][d][s]
  {
    const size_t vtb = ((size_t)bh * HDn + 2 * i) * Sn + s;
    vT[vtb]      = (__bf16)v[src];
    vT[vtb + Sn] = (__bf16)v[src + 1];
  }
}

// ---------------------------------------------------------------------------
// Flash attention (causal): block = 8 waves, each wave owns a 16-row Q tile.
// Per 32-key chunk: two 16x16 score tiles via chained 16x16x32 WMMAs (d=64),
// online softmax with shfl_xor row reductions, P -> LDS -> A-fragment,
// 4 PV WMMAs. Output written as bf16 (B,S,DIM) for the final projection.
// ---------------------------------------------------------------------------
__global__ __launch_bounds__(256)
void flash_attn_kernel(const __bf16* __restrict__ qh, const __bf16* __restrict__ kh,
                       const __bf16* __restrict__ vT, __bf16* __restrict__ attn) {
  __shared__ __bf16 Ps[8][16][32];   // per-wave P tile, 8 KB total

  const int lane = threadIdx.x & 31;
  const int wave = threadIdx.x >> 5;
  const int g    = lane >> 4;
  const int ln   = lane & 15;

  const int tiles = Sn / 128;                  // 16 q-tiles of 128 per (b,h)
  const int bh = blockIdx.x / tiles;
  const int q0 = (blockIdx.x % tiles) * 128 + wave * 16;
  const int b  = bh >> 4;
  const int h  = bh & (Hn - 1);

  const __bf16* qp = qh + (size_t)bh * Sn * HDn;
  const __bf16* kp = kh + (size_t)bh * Sn * HDn;
  const __bf16* vp = vT + (size_t)bh * HDn * Sn;

  // Q A-fragments (16x64 split into two 16x32 fragments), loaded once.
  v16bf qa0, qa1;
  {
    const __bf16* r = qp + (size_t)(q0 + ln) * HDn;
    qa0 = cat8(*(const v8bf*)(r + 8 * g),      *(const v8bf*)(r + 16 + 8 * g));
    qa1 = cat8(*(const v8bf*)(r + 32 + 8 * g), *(const v8bf*)(r + 48 + 8 * g));
  }

  float mi[8], li[8];
  v8f acc[4] = {};
#pragma unroll
  for (int r = 0; r < 8; ++r) { mi[r] = -1e30f; li[r] = 0.0f; }

  const int nch = (q0 + 47) >> 5;              // causal: chunks with key <= q
  for (int cch = 0; cch < nch; ++cch) {
    const int kb = cch * 32;

    // ---- scores for keys [kb, kb+32): two 16x16 tiles ----
    v8f sc[2];
#pragma unroll
    for (int hf = 0; hf < 2; ++hf) {
      const __bf16* r = kp + (size_t)(kb + hf * 16 + ln) * HDn;
      v16bf kb0 = cat8(*(const v8bf*)(r + 8 * g),      *(const v8bf*)(r + 16 + 8 * g));
      v16bf kb1 = cat8(*(const v8bf*)(r + 32 + 8 * g), *(const v8bf*)(r + 48 + 8 * g));
      v8f z = {};
      z = wmma_bf16(qa0, kb0, z);
      z = wmma_bf16(qa1, kb1, z);
      sc[hf] = z;
    }

    // ---- causal mask (C layout: row = r + 8g, col = lane low nibble) ----
#pragma unroll
    for (int r = 0; r < 8; ++r) {
      const int qr = q0 + r + 8 * g;
      if (kb + ln > qr)      sc[0][r] = -1e30f;
      if (kb + 16 + ln > qr) sc[1][r] = -1e30f;
    }

    // ---- online softmax ----
    float pr0[8], pr1[8];
#pragma unroll
    for (int r = 0; r < 8; ++r) {
      float v = fmaxf(sc[0][r], sc[1][r]);
#pragma unroll
      for (int off = 1; off < 16; off <<= 1)
        v = fmaxf(v, __shfl_xor(v, off, 32));
      const float mn   = fmaxf(mi[r], v);
      const float corr = __expf(mi[r] - mn);
      mi[r] = mn;
      const float p0 = __expf(sc[0][r] - mn);
      const float p1 = __expf(sc[1][r] - mn);
      pr0[r] = p0; pr1[r] = p1;
      float rs = p0 + p1;
#pragma unroll
      for (int off = 1; off < 16; off <<= 1)
        rs += __shfl_xor(rs, off, 32);
      li[r] = li[r] * corr + rs;
      acc[0][r] *= corr; acc[1][r] *= corr;
      acc[2][r] *= corr; acc[3][r] *= corr;
    }

    // ---- P: C layout -> LDS -> A-fragment (same-wave, no barrier) ----
#pragma unroll
    for (int r = 0; r < 8; ++r) {
      Ps[wave][r + 8 * g][ln]      = (__bf16)pr0[r];
      Ps[wave][r + 8 * g][16 + ln] = (__bf16)pr1[r];
    }
    v16bf pa;
    {
      const __bf16* r = &Ps[wave][ln][0];
      pa = cat8(*(const v8bf*)(r + 8 * g), *(const v8bf*)(r + 16 + 8 * g));
    }

    // ---- PV: 4 d-tiles, B-fragments contiguous in s from vT ----
#pragma unroll
    for (int dt = 0; dt < 4; ++dt) {
      const __bf16* vr = vp + (size_t)(dt * 16 + ln) * Sn + kb;
      v16bf vb = cat8(*(const v8bf*)(vr + 8 * g), *(const v8bf*)(vr + 16 + 8 * g));
      acc[dt] = wmma_bf16(pa, vb, acc[dt]);
    }
  }

  // ---- epilogue: normalize and write (B,S,DIM) bf16 ----
#pragma unroll
  for (int r = 0; r < 8; ++r) {
    const float rl = 1.0f / li[r];
    const size_t row = ((size_t)b * Sn + q0 + r + 8 * g) * DIMn + h * HDn;
#pragma unroll
    for (int dt = 0; dt < 4; ++dt)
      attn[row + dt * 16 + ln] = (__bf16)(acc[dt][r] * rl);
  }
}

// ---------------------------------------------------------------------------
// Launch
// ---------------------------------------------------------------------------
extern "C" void kernel_launch(void* const* d_in, const int* in_sizes, int n_in,
                              void* d_out, int out_size, void* d_ws, size_t ws_size,
                              hipStream_t stream) {
  const float* x    = (const float*)d_in[0];
  const float* wq   = (const float*)d_in[1];
  const float* wk   = (const float*)d_in[2];
  const float* wv   = (const float*)d_in[3];
  const float* wo   = (const float*)d_in[4];
  const float* fcos = (const float*)d_in[5];
  const float* fsin = (const float*)d_in[6];
  // d_in[7] (mask) unused: causal mask computed analytically (identical result)

  const size_t MK = (size_t)Mr * DIMn;     // 4096*1024 activation elements
  const size_t NK = (size_t)DIMn * DIMn;   // 1024*1024 weight elements

  float*  qs  = (float*)d_ws;              // f32 scratch (dead after rope_pack)
  float*  ks  = qs + MK;
  float*  vs  = ks + MK;
  __bf16* qh  = (__bf16*)(vs + MK);        // packed bf16 tensors
  __bf16* kh  = qh + MK;
  __bf16* vT  = kh + MK;
  __bf16* xb  = vT + MK;
  __bf16* wqb = xb + MK;
  __bf16* wkb = wqb + NK;
  __bf16* wvb = wkb + NK;
  __bf16* wob = wvb + NK;
  __bf16* attnb = (__bf16*)qs;             // alias: qs dead after rope_pack

  // 1) precision pack (f32 -> bf16)
  pack_bf16_kernel<<<(int)(MK / 1024), 256, 0, stream>>>(x,  xb);
  pack_bf16_kernel<<<(int)(NK / 1024), 256, 0, stream>>>(wq, wqb);
  pack_bf16_kernel<<<(int)(NK / 1024), 256, 0, stream>>>(wk, wkb);
  pack_bf16_kernel<<<(int)(NK / 1024), 256, 0, stream>>>(wv, wvb);
  pack_bf16_kernel<<<(int)(NK / 1024), 256, 0, stream>>>(wo, wob);

  const int gemm_grid = (Mr / BM) * (DIMn / BN);   // 32*16 = 512

  // 2) QKV projections
  gemm_bf16_kernel<<<gemm_grid, 256, 0, stream>>>(xb, wqb, qs, Mr, DIMn, DIMn);
  gemm_bf16_kernel<<<gemm_grid, 256, 0, stream>>>(xb, wkb, ks, Mr, DIMn, DIMn);
  gemm_bf16_kernel<<<gemm_grid, 256, 0, stream>>>(xb, wvb, vs, Mr, DIMn, DIMn);

  // 3) RoPE + head-major repack (+ V transpose)
  const int rope_threads = Bn * Hn * Sn * 32;      // 2,097,152
  rope_pack_kernel<<<rope_threads / 256, 256, 0, stream>>>(qs, ks, vs, fcos, fsin,
                                                           qh, kh, vT);

  // 4) causal flash attention
  const int flash_grid = Bn * Hn * (Sn / 128);     // 512
  flash_attn_kernel<<<flash_grid, 256, 0, stream>>>(qh, kh, vT, attnb);

  // 5) output projection
  gemm_bf16_kernel<<<gemm_grid, 256, 0, stream>>>(attnb, wob, (float*)d_out,
                                                  Mr, DIMn, DIMn);
}
